// NeuralCodebook_9070970929189
// MI455X (gfx1250) — matrix operand order repaired and checked by reading, not compiled, not run on hardware
//
#include <hip/hip_runtime.h>

// CDNA5 (gfx1250) wave32 WMMA vector-quantization nearest-neighbor.
//   argmin_k ||x-c_k||^2 == argmin_k (||c_k||^2 - 2 x.c_k)
// Score GEMM (2*N*K*D = 274 GFLOP, AI ~1900 flop/byte -> compute-bound) runs
// as bf16 WMMA with f32 accumulation. Codebook tiles are streamed into LDS
// with GLOBAL_LOAD_ASYNC_TO_LDS_B128 (ASYNCcnt) and double-buffered.

typedef __attribute__((ext_vector_type(16))) __bf16 v16bf;
typedef __attribute__((ext_vector_type(8)))  float  v8f;

namespace {
constexpr int kN = 32768;   // queries
constexpr int kK = 8192;    // codebook entries
constexpr int kD = 512;     // feature dim
constexpr int kTileK = 32;  // codebook rows per LDS buffer
constexpr int kPitch = kD + 8;  // 520 bf16 -> 1040B rows: 16B aligned, 4-bank skew
constexpr int kBufElems = kTileK * kPitch;       // bf16 elems per buffer
constexpr int kLdsBytes = 2 * kBufElems * 2;     // double buffered: 66,560 B
}

union BF16x16 { __bf16 e[16]; v16bf v; uint4 u[2]; };
union F32x8   { float  e[8];  v8f   v; };

// ---- Kernel 1: weight fp32 -> bf16 scratch copy + per-code squared norm ----
__global__ __launch_bounds__(256) void vq_prep(const float* __restrict__ w,
                                               __bf16* __restrict__ wb,
                                               float* __restrict__ csq) {
  const int lane = threadIdx.x & 31;
  const int k = blockIdx.x * 8 + (threadIdx.x >> 5);
  const float4* __restrict__ src = (const float4*)(w + (size_t)k * kD);
  __bf16* __restrict__ drow = wb + (size_t)k * kD;
  float sum = 0.f;
#pragma unroll
  for (int i = 0; i < 4; ++i) {
    float4 f = src[lane + 32 * i];
    sum = __builtin_fmaf(f.x, f.x, sum);
    sum = __builtin_fmaf(f.y, f.y, sum);
    sum = __builtin_fmaf(f.z, f.z, sum);
    sum = __builtin_fmaf(f.w, f.w, sum);
    __bf16* d = drow + (lane + 32 * i) * 4;
    d[0] = (__bf16)f.x; d[1] = (__bf16)f.y;
    d[2] = (__bf16)f.z; d[3] = (__bf16)f.w;
  }
#pragma unroll
  for (int m = 16; m >= 1; m >>= 1) sum += __shfl_xor(sum, m, 32);
  if (lane == 0) csq[k] = sum;
}

// Issue this thread's share (8 x 16B) of one kTileK x kD bf16 tile as async
// memory->LDS copies. Per-lane LDS byte address = low 32 bits of the generic
// pointer (ISA: LDS aperture keeps addr[31:0]).
__device__ __forceinline__ void stage_async(const __bf16* __restrict__ wb,
                                            __bf16* tileBase, int kt, int tid) {
#pragma unroll
  for (int i = 0; i < (kTileK * (kD / 8)) / 256; ++i) {  // 8 chunks/thread
    int chunk = tid + 256 * i;
    int r  = chunk >> 6;            // kD/8 == 64 16B-chunks per row
    int c8 = (chunk & 63) * 8;
    unsigned ldsOff = (unsigned)(uintptr_t)(tileBase + r * kPitch + c8);
    const void* g = (const void*)(wb + (size_t)(kt + r) * kD + c8);
    asm volatile("global_load_async_to_lds_b128 %0, %1, off"
                 :: "v"(ldsOff), "v"(g) : "memory");
  }
}

// ---- Kernel 2: fused score GEMM + running argmin --------------------------
// 8 waves/block; each wave owns a 16-row x tile resident as 16 A fragments.
// Codebook streamed through a double-buffered LDS tile via async copies.
__global__ __launch_bounds__(256) void vq_argmin(const float* __restrict__ x,
                                                 const __bf16* __restrict__ wb,
                                                 const float* __restrict__ csq,
                                                 int* __restrict__ bestIdx) {
  extern __shared__ __bf16 tile[];  // [2][kTileK][kPitch]
  const int tid  = threadIdx.x;
  const int lane = tid & 31;
  const int wave = tid >> 5;
  const int hi   = lane >> 4;
  const int col  = lane & 15;
  const int rowBase = blockIdx.x * 128 + wave * 16;

  // Preload A fragments (16x32 per fragment, K order matching B):
  // lane<16 holds K=0..15; lane>=16 holds K=16..31; row M=lane&15.
  BF16x16 afrag[16];
  {
    const float* __restrict__ xr = x + (size_t)(rowBase + col) * kD;
#pragma unroll
    for (int dc = 0; dc < 16; ++dc) {
      const float4* p0 = (const float4*)(xr + dc * 32 + hi * 16);
      const float4* p1 = (const float4*)(xr + dc * 32 + hi * 16 + 8);
      float4 f0 = p0[0], f1 = p0[1], f2 = p1[0], f3 = p1[1];
      float t[16] = {f0.x, f0.y, f0.z, f0.w, f1.x, f1.y, f1.z, f1.w,
                     f2.x, f2.y, f2.z, f2.w, f3.x, f3.y, f3.z, f3.w};
#pragma unroll
      for (int e = 0; e < 16; ++e) afrag[dc].e[e] = (__bf16)t[e];
    }
  }

  float best[8];
  int   bidx[8];
#pragma unroll
  for (int v = 0; v < 8; ++v) { best[v] = 3.402823466e38f; bidx[v] = 0; }

  constexpr int nIter = kK / kTileK;
  stage_async(wb, tile, /*kt=*/0, tid);  // prologue fill of buffer 0

  for (int it = 0; it < nIter; ++it) {
    const int p  = it & 1;
    const int kt = it * kTileK;
    asm volatile("s_wait_asynccnt 0x0" ::: "memory");  // my async copies done
    __syncthreads();                                    // everyone's done
    if (it + 1 < nIter)  // fill other buffer while we compute on this one
      stage_async(wb, tile + (1 - p) * kBufElems, kt + kTileK, tid);

    __bf16* base = tile + p * kBufElems;
    // B fragment (32x16), contiguous K order as in A: lane holds column n=lane&15, elements
    // 0..15 = K(hi*16..hi*16+15) -> one contiguous 32B LDS read per fragment.
    const __bf16* bp0 = base + col * kPitch + hi * 16;         // codes kt+0..15
    const __bf16* bp1 = base + (16 + col) * kPitch + hi * 16;  // codes kt+16..31
    const float cs0 = csq[kt + col];
    const float cs1 = csq[kt + 16 + col];

    v8f acc0 = {0.f, 0.f, 0.f, 0.f, 0.f, 0.f, 0.f, 0.f};
    v8f acc1 = {0.f, 0.f, 0.f, 0.f, 0.f, 0.f, 0.f, 0.f};
    BF16x16 b0, b1, n0, n1;
    b0.u[0] = *(const uint4*)(bp0);      b0.u[1] = *(const uint4*)(bp0 + 8);
    b1.u[0] = *(const uint4*)(bp1);      b1.u[1] = *(const uint4*)(bp1 + 8);
#pragma unroll
    for (int dc = 0; dc < 16; ++dc) {
      if (dc + 1 < 16) {  // software pipeline: next-dc B loads before WMMAs
        n0.u[0] = *(const uint4*)(bp0 + (dc + 1) * 32);
        n0.u[1] = *(const uint4*)(bp0 + (dc + 1) * 32 + 8);
        n1.u[0] = *(const uint4*)(bp1 + (dc + 1) * 32);
        n1.u[1] = *(const uint4*)(bp1 + (dc + 1) * 32 + 8);
      }
      acc0 = __builtin_amdgcn_wmma_f32_16x16x32_bf16(
          false, afrag[dc].v, false, b0.v, (short)0, acc0, false, false);
      acc1 = __builtin_amdgcn_wmma_f32_16x16x32_bf16(
          false, afrag[dc].v, false, b1.v, (short)0, acc1, false, false);
      b0 = n0; b1 = n1;
    }

    // C layout: lane l, slot v -> row M = v + 8*hi, column N = l&15.
    F32x8 r0; r0.v = acc0;
    F32x8 r1; r1.v = acc1;
#pragma unroll
    for (int v = 0; v < 8; ++v) {
      float s0 = __builtin_fmaf(-2.f, r0.e[v], cs0);
      float s1 = __builtin_fmaf(-2.f, r1.e[v], cs1);
      if (s0 < best[v]) { best[v] = s0; bidx[v] = kt + col; }
      if (s1 < best[v]) { best[v] = s1; bidx[v] = kt + 16 + col; }
    }
  }

  // Cross-lane argmin butterfly within each 16-lane half (rows 0-7 live in
  // lanes 0-15, rows 8-15 in lanes 16-31). Tie-break toward lower index.
#pragma unroll
  for (int v = 0; v < 8; ++v) {
    float s = best[v];
    int   i = bidx[v];
#pragma unroll
    for (int m = 8; m >= 1; m >>= 1) {
      float s2 = __shfl_xor(s, m, 32);
      int   i2 = __shfl_xor(i, m, 32);
      if (s2 < s || (s2 == s && i2 < i)) { s = s2; i = i2; }
    }
    if (col == 0) bestIdx[rowBase + hi * 8 + v] = i;
  }
}

// ---- Kernel 3: gather fp32 codebook rows into the output ------------------
__global__ __launch_bounds__(128) void vq_gather(const float* __restrict__ w,
                                                 const int* __restrict__ bestIdx,
                                                 float* __restrict__ out) {
  const int row = blockIdx.x;
  const int c = bestIdx[row];
  const float4* __restrict__ src = (const float4*)(w + (size_t)c * kD);
  float4* __restrict__ dst = (float4*)(out + (size_t)row * kD);
  dst[threadIdx.x] = src[threadIdx.x];
}

extern "C" void kernel_launch(void* const* d_in, const int* in_sizes, int n_in,
                              void* d_out, int out_size, void* d_ws, size_t ws_size,
                              hipStream_t stream) {
  (void)in_sizes; (void)n_in; (void)out_size; (void)ws_size;
  const float* x = (const float*)d_in[0];
  const float* w = (const float*)d_in[1];
  float* out = (float*)d_out;

  // Scratch layout: csq (32 KB) | best indices (128 KB) | bf16 weight (8 MB)
  float*  csq  = (float*)d_ws;
  int*    best = (int*)(csq + kK);
  __bf16* wb   = (__bf16*)(best + kN);

  vq_prep  <<<kK / 8,   256, 0,         stream>>>(w, wb, csq);
  vq_argmin<<<kN / 128, 256, kLdsBytes, stream>>>(x, wb, csq, best);
  vq_gather<<<kN,       128, 0,         stream>>>(w, best, out);
}
